// Model_78812649882222
// MI455X (gfx1250) — compile-verified, matrix operand order checked by reading, not run on hardware
//
#include <hip/hip_runtime.h>
#include <hip/hip_bf16.h>

// ---------------------------------------------------------------------------
// CDNA5 / gfx1250 implementation. wave32, WMMA bf16 16x16x32, f32 accumulate.
// LDS tiles are stored fragment-major so fragments load as ds_load_b128 pairs.
// ---------------------------------------------------------------------------

typedef __attribute__((ext_vector_type(16))) __bf16 v16bf;
typedef __attribute__((ext_vector_type(8)))  float  v8f;

#define B_    64
#define T_    1024
#define N_    21
#define V_    20          // N-1 covariates
#define PL_   16
#define PN_   64
#define DM_   512
#define DFF_  1024
#define PRED_ 720
#define TOPKK 8192        // PATCH_NUM*(N-1)*PATCH_NUM*0.1
#define ATTL  (V_*PN_)    // 1280 attention cols
#define EPSF  1e-5f

// ------------------------------ helpers ------------------------------------

__device__ __forceinline__ unsigned short f32_to_bf16u(float f) {
  unsigned int u = __float_as_uint(f);
  u += 0x7FFFu + ((u >> 16) & 1u);            // round-to-nearest-even
  return (unsigned short)(u >> 16);
}

__device__ __forceinline__ void block_reduce2(float& s, float& q,
                                              float* sh1, float* sh2) {
  int tid = threadIdx.x;
  sh1[tid] = s; sh2[tid] = q; __syncthreads();
  for (int o = 128; o > 0; o >>= 1) {
    if (tid < o) { sh1[tid] += sh1[tid + o]; sh2[tid] += sh2[tid + o]; }
    __syncthreads();
  }
  s = sh1[0]; q = sh2[0];
}

__global__ void zero_u32_kernel(unsigned int* p, long n) {
  long i = (long)blockIdx.x * blockDim.x + threadIdx.x;
  if (i < n) p[i] = 0u;
}

__global__ void cvt_bf16_kernel(const float* __restrict__ x,
                                unsigned short* __restrict__ y, long n) {
  long i = (long)blockIdx.x * blockDim.x + threadIdx.x;
  if (i < n) y[i] = f32_to_bf16u(x[i]);
}

// -------------------- per-channel normalization (RevIN-ish) -----------------
// x_enc (B,T,N) -> xn (B,N,T); target-channel mean/std(ddof=1) saved.

__global__ void channel_norm_kernel(const float* __restrict__ x_enc,
                                    float* __restrict__ xn,
                                    float* __restrict__ tstats) {
  __shared__ float sh1[256], sh2[256];
  int b = blockIdx.x / N_, n = blockIdx.x % N_;
  int tid = threadIdx.x;
  float s = 0.f, q = 0.f;
  for (int t = tid; t < T_; t += 256) {
    float v = x_enc[((long)b * T_ + t) * N_ + n];
    s += v; q += v * v;
  }
  block_reduce2(s, q, sh1, sh2);
  float mean = s / T_;
  float var  = fmaxf((q - T_ * mean * mean) / (T_ - 1), 0.f);
  float stdd = sqrtf(var);
  float inv  = 1.f / (stdd + EPSF);
  for (int t = tid; t < T_; t += 256) {
    float v = x_enc[((long)b * T_ + t) * N_ + n];
    xn[((long)b * N_ + n) * T_ + t] = (v - mean) * inv;
  }
  if (n == N_ - 1 && tid == 0) { tstats[b] = mean; tstats[B_ + b] = stdd; }
}

// --------------- per-batch whole-region LayerNorm (two kernels) -------------

__global__ void region_stats_kernel(const float* __restrict__ x, long xstride_b,
                                    long L, float* __restrict__ acc) {
  __shared__ float sh1[256], sh2[256];
  int b = blockIdx.y;
  const float* xb = x + (long)b * xstride_b;
  float s = 0.f, q = 0.f;
  for (long i = (long)blockIdx.x * 256 + threadIdx.x; i < L;
       i += (long)gridDim.x * 256) {
    float v = xb[i]; s += v; q += v * v;
  }
  block_reduce2(s, q, sh1, sh2);
  if (threadIdx.x == 0) { atomicAdd(&acc[b], s); atomicAdd(&acc[B_ + b], q); }
}

// y = (optional relu)((x - mu)*rsqrt(var+eps)); ln gamma==1, beta==0 in ref.
__global__ void ln_apply_kernel(const float* __restrict__ x, long xstride_b,
                                const float* __restrict__ acc, long L, int relu,
                                float* __restrict__ yf,
                                unsigned short* __restrict__ yb) {
  int b = blockIdx.y;
  const float* xb = x + (long)b * xstride_b;
  float mean = acc[b] / (float)L;
  float var  = fmaxf(acc[B_ + b] / (float)L - mean * mean, 0.f);
  float inv  = rsqrtf(var + EPSF);
  for (long i = (long)blockIdx.x * 256 + threadIdx.x; i < L;
       i += (long)gridDim.x * 256) {
    float v = (xb[i] - mean) * inv;
    if (relu) v = fmaxf(v, 0.f);
    long o = (long)b * L + i;
    if (yf) yf[o] = v;
    if (yb) yb[o] = f32_to_bf16u(v);
  }
}

// ------------------------------ WMMA GEMM -----------------------------------
// C[M,N] = act(scale*(A[M,K] @ B) + bias), A/B bf16, f32 accumulate.
// TRANSB: B stored (N,K) row-major (used for Q @ K^T).
// Batched via grid.z (strides sAb/sBb/sCb) or split-K (ksplit>1, atomicAcc).
//
// LDS layout is fragment-major: per 16x16x32 hardware tile, lane l's 16 bf16
// A/B fragment elements occupy one contiguous 32-byte slot, so a fragment
// loads with two ds_load_b128 and zero packing VALU.
// Fragment element mapping (ISA 16-bit A 16x32):
//   lane = half*16 + r ; u[0..7] = K half*8 .. half*8+7 ; u[8..15] = +16.
//   -> for K index k: half=(k>>3)&1, pos=(k&7)+8*(k>>4).

#define BM 64
#define BN 128
#define KT 32

template <int TRANSB>
__global__ __launch_bounds__(256) void wmma_gemm_kernel(
    const unsigned short* __restrict__ A, const unsigned short* __restrict__ Bm,
    const float* __restrict__ bias, float* __restrict__ Cf,
    unsigned short* __restrict__ Cb, int M, int N, int K, long sAb, long sBb,
    long sCb, float scale, int relu, int ksplit, int atomicAcc) {
  // 4 A-tiles * 32 lanes * 16 bf16 = 4 KB ; 8 B-tiles -> 8 KB
  __shared__ alignas(32) unsigned short lA[4 * 32 * 16];
  __shared__ alignas(32) unsigned short lB[8 * 32 * 16];

  int bz = blockIdx.z;
  const unsigned short* Ab = A;
  const unsigned short* Bb = Bm;
  long cOff = 0;
  int kIters = (K + KT - 1) / KT, k0i = 0, k1i = kIters;
  if (ksplit > 1) {
    int per = (kIters + ksplit - 1) / ksplit;
    k0i = bz * per;
    k1i = (k0i + per < kIters) ? k0i + per : kIters;
  } else {
    Ab += (long)bz * sAb; Bb += (long)bz * sBb; cOff = (long)bz * sCb;
  }

  int m0 = blockIdx.y * BM, n0 = blockIdx.x * BN;
  int tid = threadIdx.x, lane = tid & 31, wave = tid >> 5;
  int half = lane >> 4, r = lane & 15;
  int tr = wave & 3;            // this wave's 16-row tile within BM
  int tcg = (wave >> 2) * 4;    // first of 4 consecutive 16-col tiles

  v8f acc[4] = {};

  // A staging decomposition: tid -> (row rr, 8-wide K chunk c8)
  const int a_rr = tid >> 2, a_c8 = tid & 3;
  const int a_t = a_rr >> 4, a_r = a_rr & 15;
  const int a_half = a_c8 & 1, a_kgrp = a_c8 >> 1;
  unsigned short* a_dst =
      &lA[((a_t * 32 + a_half * 16 + a_r) * 16) + a_kgrp * 8];

  for (int kt = k0i; kt < k1i; ++kt) {
    int kb = kt * KT;

    // ---- stage A tile (BM x KT): one b128 global load + one b128 LDS store
    {
      int gm = m0 + a_rr, gk = kb + a_c8 * 8;
      uint4 val = {0u, 0u, 0u, 0u};
      if (gm < M && gk + 8 <= K)
        val = *reinterpret_cast<const uint4*>(&Ab[(long)gm * K + gk]);
      *reinterpret_cast<uint4*>(a_dst) = val;
    }

    // ---- stage B tile (KT x BN)
    if (TRANSB) {
      // B stored (N,K): 8 consecutive K of one column -> one contiguous slot
      for (int it = tid; it < BN * 4; it += 256) {
        int nn = it >> 2, c8 = it & 3;
        int gn = n0 + nn, gk = kb + c8 * 8;
        uint4 val = {0u, 0u, 0u, 0u};
        if (gn < N && gk + 8 <= K)
          val = *reinterpret_cast<const uint4*>(&Bb[(long)gn * K + gk]);
        int tc = nn >> 4, rr = nn & 15;
        int hf = c8 & 1, kg = c8 >> 1;
        *reinterpret_cast<uint4*>(
            &lB[((tc * 32 + hf * 16 + rr) * 16) + kg * 8]) = val;
      }
    } else {
      // B stored (K,N): load two adjacent K rows (coalesced b128), scatter as
      // packed K-pairs (adjacent pos -> aligned b32 LDS stores).
      int kp = tid >> 4;            // 0..15 -> k0 = 2*kp
      int n8 = tid & 15;            // 8-wide column chunk
      int k0 = kp * 2;
      int gk0 = kb + k0;
      int gn = n0 + n8 * 8;
      union { uint4 u4; unsigned short us[8]; } r0, r1;
      r0.u4 = (uint4){0u, 0u, 0u, 0u};
      r1.u4 = (uint4){0u, 0u, 0u, 0u};
      if (gn + 8 <= N) {
        if (gk0 < K)
          r0.u4 = *reinterpret_cast<const uint4*>(&Bb[(long)gk0 * N + gn]);
        if (gk0 + 1 < K)
          r1.u4 = *reinterpret_cast<const uint4*>(&Bb[(long)(gk0 + 1) * N + gn]);
      }
      int hf = (k0 >> 3) & 1;
      int pos = (k0 & 7) + 8 * (k0 >> 4);   // even -> 4B aligned
#pragma unroll
      for (int i = 0; i < 8; ++i) {
        int nn = n8 * 8 + i;
        int tc = nn >> 4, rr = nn & 15;
        unsigned int pk =
            (unsigned int)r0.us[i] | ((unsigned int)r1.us[i] << 16);
        *reinterpret_cast<unsigned int*>(
            &lB[((tc * 32 + hf * 16 + rr) * 16) + pos]) = pk;
      }
    }

    // prefetch next A K-tile into cache (gfx1250 global_prefetch path)
    if (kt + 1 < k1i) {
      int gm = m0 + a_rr;
      if (gm < M && (kb + KT + a_c8 * 8 + 8) <= K)
        __builtin_prefetch(&Ab[(long)gm * K + kb + KT + a_c8 * 8], 0, 1);
    }
    __syncthreads();

    // ---- fragments: contiguous 32B slots -> 2x ds_load_b128 each
    v16bf av = *reinterpret_cast<const v16bf*>(&lA[(tr * 32 + lane) * 16]);
#pragma unroll
    for (int tc = 0; tc < 4; ++tc) {
      v16bf bv = *reinterpret_cast<const v16bf*>(
          &lB[((tcg + tc) * 32 + lane) * 16]);
      acc[tc] = __builtin_amdgcn_wmma_f32_16x16x32_bf16(
          false, av, false, bv, (short)0, acc[tc], false, false);
    }
    __syncthreads();
  }

  // epilogue: C layout -> VGPR j: M = tile + half*8 + j, N = tile + r
#pragma unroll
  for (int tc = 0; tc < 4; ++tc) {
    int col = n0 + (tcg + tc) * 16 + r;
    if (col >= N) continue;
#pragma unroll
    for (int j = 0; j < 8; ++j) {
      int row = m0 + tr * 16 + half * 8 + j;
      if (row >= M) continue;
      float v = acc[tc][j] * scale;
      long idx = cOff + (long)row * N + col;
      if (atomicAcc) {
        atomicAdd(&Cf[idx], v);
      } else {
        if (bias) v += bias[col];
        if (relu) v = fmaxf(v, 0.f);
        if (Cf) Cf[idx] = v;
        if (Cb) Cb[idx] = f32_to_bf16u(v);
      }
    }
  }
}

// --------------------- paired softmax (attn & nattn) ------------------------

__global__ void softmax_pair_kernel(const float* __restrict__ scores,
                                    float* __restrict__ attn,
                                    float* __restrict__ nattn) {
  __shared__ float sh1[256], sh2[256];
  long row = blockIdx.x;
  const float* s = scores + row * ATTL;
  int tid = threadIdx.x;
  float mx = -3.4e38f, mn = 3.4e38f;
  for (int i = tid; i < ATTL; i += 256) {
    float v = s[i]; mx = fmaxf(mx, v); mn = fminf(mn, v);
  }
  sh1[tid] = mx; sh2[tid] = mn; __syncthreads();
  for (int o = 128; o > 0; o >>= 1) {
    if (tid < o) {
      sh1[tid] = fmaxf(sh1[tid], sh1[tid + o]);
      sh2[tid] = fminf(sh2[tid], sh2[tid + o]);
    }
    __syncthreads();
  }
  mx = sh1[0]; mn = sh2[0]; __syncthreads();
  float se = 0.f, sn = 0.f;
  for (int i = tid; i < ATTL; i += 256) {
    float v = s[i]; se += expf(v - mx); sn += expf(mn - v);
  }
  block_reduce2(se, sn, sh1, sh2);
  float ise = 1.f / se, isn = 1.f / sn;
  for (int i = tid; i < ATTL; i += 256) {
    float v = s[i];
    attn[row * ATTL + i]  = expf(v - mx) * ise;
    nattn[row * ATTL + i] = expf(mn - v) * isn;
  }
}

// ------------------ top-k threshold: 4-pass radix select --------------------
// softmax outputs are >=0 so float ordering == uint-bit ordering.

__global__ void topk_kernel(const float* __restrict__ attn,
                            unsigned int* __restrict__ thr,
                            int* __restrict__ ties) {
  const long L = (long)PN_ * ATTL;  // 81920
  int b = blockIdx.x, tid = threadIdx.x;
  const float* a = attn + (long)b * L;
  __shared__ unsigned int hist[256];
  __shared__ unsigned int s_prefix;
  __shared__ int s_want;
  if (tid == 0) { s_prefix = 0u; s_want = TOPKK; }
  __syncthreads();
  for (int pass = 0; pass < 4; ++pass) {
    int shift = 24 - 8 * pass;
    unsigned int pmask = (pass == 0) ? 0u : (0xFFFFFFFFu << (32 - 8 * pass));
    hist[tid] = 0; __syncthreads();
    unsigned int pref = s_prefix;
    for (long i = tid; i < L; i += 256) {
      unsigned int u = __float_as_uint(a[i]);
      if ((u & pmask) == pref) atomicAdd(&hist[(u >> shift) & 255], 1u);
    }
    __syncthreads();
    if (tid == 0) {
      unsigned int cum = 0; int want = s_want; int sel = 0;
      for (int d = 255; d >= 0; --d) {
        if (cum + hist[d] >= (unsigned)want) { sel = d; s_want = want - (int)cum; break; }
        cum += hist[d];
      }
      s_prefix = pref | ((unsigned)sel << shift);
    }
    __syncthreads();
  }
  if (tid == 0) { thr[b] = s_prefix; ties[b] = s_want; }
}

// maskI*attn and (1-maskI)*nattn as bf16 GEMM inputs (tie pick via counter)
__global__ void apply_mask_kernel(const float* __restrict__ attn,
                                  const float* __restrict__ nattn,
                                  const unsigned int* __restrict__ thr,
                                  const int* __restrict__ ties,
                                  int* __restrict__ tiecnt,
                                  unsigned short* __restrict__ AI,
                                  unsigned short* __restrict__ AV) {
  const long L = (long)PN_ * ATTL;
  long i = (long)blockIdx.x * 256 + threadIdx.x;
  if (i >= (long)B_ * L) return;
  int b = (int)(i / L);
  unsigned int u = __float_as_uint(attn[i]);
  int sel;
  if (u > thr[b]) sel = 1;
  else if (u == thr[b]) sel = (atomicAdd(&tiecnt[b], 1) < ties[b]);
  else sel = 0;
  AI[i] = f32_to_bf16u(sel ? attn[i] : 0.f);
  AV[i] = f32_to_bf16u(sel ? 0.f : nattn[i]);
}

// ------------- sigmoid gating + assemble 3 head inputs (bf16) ---------------

__global__ void fuse_gate_kernel(const float* __restrict__ InterI,
                                 const float* __restrict__ InterV,
                                 const float* __restrict__ Md,
                                 const float* __restrict__ End,
                                 const float* __restrict__ alpha,
                                 unsigned short* __restrict__ H) {
  const long TOT = (long)B_ * PN_ * DM_;
  long i = (long)blockIdx.x * 256 + threadIdx.x;
  if (i >= TOT) return;
  float m = Md[i];
  float msha = 1.f / (1.f + expf(-m));   // TAU = 1
  float mspe = 1.f / (1.f + expf(m));
  float ii = InterI[i];
  float ri = msha * ii, rv = mspe * ii;
  float a = 1.f / (1.f + expf(-alpha[0]));
  float z = a * End[i] + (1.f - a) * ri;
  long b = i >> 15;            // / 32768
  long off = i & 32767;
  const long RL = (long)PN_ * DM_;
  H[((long)0 * B_ + b) * RL + off] = f32_to_bf16u(z);
  H[((long)1 * B_ + b) * RL + off] = f32_to_bf16u(z + InterV[i]);
  H[((long)2 * B_ + b) * RL + off] = f32_to_bf16u(z + rv);
}

// ----------------------- head: +b1, row LN, relu ----------------------------

__global__ void head_ln_kernel(const float* __restrict__ E1,
                               const float* __restrict__ b1,
                               const float* __restrict__ g,
                               const float* __restrict__ bb,
                               unsigned short* __restrict__ out) {
  __shared__ float sh1[256], sh2[256];
  int rI = blockIdx.x, tid = threadIdx.x;
  const float* xr = E1 + (long)rI * DFF_;
  float s = 0.f, q = 0.f;
  for (int c = tid; c < DFF_; c += 256) {
    float v = xr[c] + b1[c]; s += v; q += v * v;
  }
  block_reduce2(s, q, sh1, sh2);
  float mean = s / DFF_;
  float var  = fmaxf(q / DFF_ - mean * mean, 0.f);
  float inv  = rsqrtf(var + EPSF);
  for (int c = tid; c < DFF_; c += 256) {
    float v = ((xr[c] + b1[c]) - mean) * inv * g[c] + bb[c];
    out[(long)rI * DFF_ + c] = f32_to_bf16u(fmaxf(v, 0.f));
  }
}

__global__ void final_out_kernel(const float* __restrict__ E2,
                                 const float* __restrict__ tstats,
                                 float* __restrict__ out) {
  long i = (long)blockIdx.x * 256 + threadIdx.x;
  const long TOT = (long)3 * B_ * PRED_;
  if (i >= TOT) return;
  int row = (int)(i / PRED_);      // row = s*B + b
  int b = row % B_;
  out[i] = E2[i] * tstats[B_ + b] + tstats[b];
}

// ------------------------------ host side -----------------------------------

static constexpr size_t AL(size_t x) { return (x + 255) & ~(size_t)255; }

// workspace layout (bytes); peak ~770 MB with BIG-region reuse
constexpr size_t OFF_XN    = 0;
constexpr size_t OFF_TST   = AL(OFF_XN + (size_t)B_ * N_ * T_ * 4);
constexpr size_t OFF_ACC   = AL(OFF_TST + 2 * B_ * 4);
constexpr size_t OFF_THR   = AL(OFF_ACC + 2 * B_ * 4);
constexpr size_t OFF_TIES  = AL(OFF_THR + B_ * 4);
constexpr size_t OFF_TIECT = AL(OFF_TIES + B_ * 4);
constexpr size_t OFF_W1C   = AL(OFF_TIECT + B_ * 4);
constexpr size_t OFF_W2C   = AL(OFF_W1C + (size_t)PL_ * DFF_ * 2);
constexpr size_t OFF_W1T   = AL(OFF_W2C + (size_t)DFF_ * DM_ * 2);
constexpr size_t OFF_W2T   = AL(OFF_W1T + (size_t)PL_ * DFF_ * 2);
constexpr size_t OFF_QW    = AL(OFF_W2T + (size_t)DFF_ * DM_ * 2);
constexpr size_t OFF_KW    = AL(OFF_QW + (size_t)DM_ * DM_ * 2);
constexpr size_t OFF_VW    = AL(OFF_KW + (size_t)DM_ * DM_ * 2);
constexpr size_t OFF_DW    = AL(OFF_VW + (size_t)DM_ * DM_ * 2);
constexpr size_t OFF_HW1   = AL(OFF_DW + (size_t)DM_ * DM_ * 2);
constexpr size_t OFF_HW2   = AL(OFF_HW1 + (size_t)PN_ * DM_ * DFF_ * 2);
constexpr size_t OFF_A1C   = AL(OFF_HW2 + (size_t)DFF_ * PRED_ * 2);
constexpr size_t OFF_A1T   = AL(OFF_A1C + (size_t)B_ * V_ * T_ * 2);
constexpr size_t OFF_BIG1  = AL(OFF_A1T + (size_t)B_ * T_ * 2);          // 336 MB f32
constexpr size_t OFF_BIG2  = AL(OFF_BIG1 + (size_t)B_ * V_ * PN_ * DFF_ * 4); // 168 MB bf16
constexpr size_t OFF_C1T   = AL(OFF_BIG2 + (size_t)B_ * V_ * PN_ * DFF_ * 2);
constexpr size_t OFF_A2T   = AL(OFF_C1T + (size_t)B_ * PN_ * DFF_ * 4);
constexpr size_t OFF_C2T   = AL(OFF_A2T + (size_t)B_ * PN_ * DFF_ * 2);
constexpr size_t OFF_ENDF  = AL(OFF_C2T + (size_t)B_ * PN_ * DM_ * 4);
constexpr size_t OFF_ENDB  = AL(OFF_ENDF + (size_t)B_ * PN_ * DM_ * 4);
constexpr size_t OFF_QB    = AL(OFF_ENDB + (size_t)B_ * PN_ * DM_ * 2);
constexpr size_t OFF_SC    = AL(OFF_QB + (size_t)B_ * PN_ * DM_ * 2);
constexpr size_t OFF_ATTN  = AL(OFF_SC + (size_t)B_ * PN_ * ATTL * 4);
constexpr size_t OFF_NATT  = AL(OFF_ATTN + (size_t)B_ * PN_ * ATTL * 4);
constexpr size_t OFF_AIB   = AL(OFF_NATT + (size_t)B_ * PN_ * ATTL * 4);
constexpr size_t OFF_AVB   = AL(OFF_AIB + (size_t)B_ * PN_ * ATTL * 2);
constexpr size_t OFF_II    = AL(OFF_AVB + (size_t)B_ * PN_ * ATTL * 2);
constexpr size_t OFF_IIB   = AL(OFF_II + (size_t)B_ * PN_ * DM_ * 4);
constexpr size_t OFF_IV    = AL(OFF_IIB + (size_t)B_ * PN_ * DM_ * 2);
constexpr size_t OFF_MD    = AL(OFF_IV + (size_t)B_ * PN_ * DM_ * 4);
constexpr size_t OFF_H     = AL(OFF_MD + (size_t)B_ * PN_ * DM_ * 4);
constexpr size_t OFF_E1    = AL(OFF_H + (size_t)3 * B_ * PN_ * DM_ * 2);
constexpr size_t OFF_E1B   = AL(OFF_E1 + (size_t)3 * B_ * DFF_ * 4);
constexpr size_t OFF_E2    = AL(OFF_E1B + (size_t)3 * B_ * DFF_ * 2);

// aliases inside the two BIG regions (lifetimes are disjoint)
constexpr size_t OFF_C1C  = OFF_BIG1;                                  // f32 cov D_FF act
constexpr size_t OFF_C2C  = OFF_BIG1;                                  // f32 cov D_MODEL act
constexpr size_t OFF_KB   = OFF_BIG1;                                  // bf16 K
constexpr size_t OFF_VB   = OFF_BIG1 + (size_t)B_ * ATTL * DM_ * 2;    // bf16 V
constexpr size_t OFF_A2C  = OFF_BIG2;                                  // bf16 cov D_FF act
constexpr size_t OFF_EXOB = OFF_BIG2;                                  // bf16 Exo

extern "C" void kernel_launch(void* const* d_in, const int* in_sizes, int n_in,
                              void* d_out, int out_size, void* d_ws,
                              size_t ws_size, hipStream_t stream) {
  (void)in_sizes; (void)n_in; (void)out_size; (void)ws_size;
  // flattened param order (insertion order of nested dicts)
  const float* x_enc  = (const float*)d_in[0];
  const float* cW1    = (const float*)d_in[6];
  const float* cb1    = (const float*)d_in[7];
  const float* cW2    = (const float*)d_in[10];
  const float* cb2    = (const float*)d_in[11];
  const float* tW1    = (const float*)d_in[16];
  const float* tb1    = (const float*)d_in[17];
  const float* tW2    = (const float*)d_in[20];
  const float* tb2    = (const float*)d_in[21];
  const float* qW     = (const float*)d_in[24];
  const float* qb     = (const float*)d_in[25];
  const float* kW     = (const float*)d_in[26];
  const float* kb     = (const float*)d_in[27];
  const float* vW     = (const float*)d_in[28];
  const float* vb     = (const float*)d_in[29];
  const float* dW     = (const float*)d_in[30];
  const float* db     = (const float*)d_in[31];
  const float* alpha  = (const float*)d_in[32];
  const float* hW1    = (const float*)d_in[33];
  const float* hb1    = (const float*)d_in[34];
  const float* hlg    = (const float*)d_in[35];
  const float* hlb    = (const float*)d_in[36];
  const float* hW2    = (const float*)d_in[37];
  const float* hb2    = (const float*)d_in[38];

  char* W = (char*)d_ws;
  float*          XN   = (float*)(W + OFF_XN);
  float*          TST  = (float*)(W + OFF_TST);
  float*          ACC  = (float*)(W + OFF_ACC);
  unsigned int*   THR  = (unsigned int*)(W + OFF_THR);
  int*            TIES = (int*)(W + OFF_TIES);
  int*            TIEC = (int*)(W + OFF_TIECT);
  unsigned short* W1C  = (unsigned short*)(W + OFF_W1C);
  unsigned short* W2C  = (unsigned short*)(W + OFF_W2C);
  unsigned short* W1T  = (unsigned short*)(W + OFF_W1T);
  unsigned short* W2T  = (unsigned short*)(W + OFF_W2T);
  unsigned short* QW   = (unsigned short*)(W + OFF_QW);
  unsigned short* KW   = (unsigned short*)(W + OFF_KW);
  unsigned short* VW   = (unsigned short*)(W + OFF_VW);
  unsigned short* DW   = (unsigned short*)(W + OFF_DW);
  unsigned short* HW1  = (unsigned short*)(W + OFF_HW1);
  unsigned short* HW2  = (unsigned short*)(W + OFF_HW2);
  unsigned short* A1C  = (unsigned short*)(W + OFF_A1C);
  unsigned short* A1T  = (unsigned short*)(W + OFF_A1T);
  float*          C1C  = (float*)(W + OFF_C1C);
  float*          C2C  = (float*)(W + OFF_C2C);
  unsigned short* KB   = (unsigned short*)(W + OFF_KB);
  unsigned short* VB   = (unsigned short*)(W + OFF_VB);
  unsigned short* A2C  = (unsigned short*)(W + OFF_A2C);
  unsigned short* EXOB = (unsigned short*)(W + OFF_EXOB);
  float*          C1T  = (float*)(W + OFF_C1T);
  unsigned short* A2T  = (unsigned short*)(W + OFF_A2T);
  float*          C2T  = (float*)(W + OFF_C2T);
  float*          ENDF = (float*)(W + OFF_ENDF);
  unsigned short* ENDB = (unsigned short*)(W + OFF_ENDB);
  unsigned short* QB   = (unsigned short*)(W + OFF_QB);
  float*          SC   = (float*)(W + OFF_SC);
  float*          ATTN = (float*)(W + OFF_ATTN);
  float*          NATT = (float*)(W + OFF_NATT);
  unsigned short* AIB  = (unsigned short*)(W + OFF_AIB);
  unsigned short* AVB  = (unsigned short*)(W + OFF_AVB);
  float*          II   = (float*)(W + OFF_II);
  unsigned short* IIB  = (unsigned short*)(W + OFF_IIB);
  float*          IV   = (float*)(W + OFF_IV);
  float*          MD   = (float*)(W + OFF_MD);
  unsigned short* H    = (unsigned short*)(W + OFF_H);
  float*          E1   = (float*)(W + OFF_E1);
  unsigned short* E1B  = (unsigned short*)(W + OFF_E1B);
  float*          E2   = (float*)(W + OFF_E2);

  auto zero = [&](void* p, long n_u32) {
    zero_u32_kernel<<<(unsigned)((n_u32 + 255) / 256), 256, 0, stream>>>(
        (unsigned int*)p, n_u32);
  };
  auto cvt = [&](const float* x, unsigned short* y, long n) {
    cvt_bf16_kernel<<<(unsigned)((n + 255) / 256), 256, 0, stream>>>(x, y, n);
  };
  auto gemm = [&](const unsigned short* A, const unsigned short* Bp,
                  const float* bias, float* Cf, unsigned short* Cb, int M, int N,
                  int K, long sA, long sB, long sC, int batch, float scale,
                  int relu, int transB, int ksplit, int atomicAcc) {
    dim3 grid((N + BN - 1) / BN, (M + BM - 1) / BM,
              (ksplit > 1) ? ksplit : batch);
    if (transB)
      wmma_gemm_kernel<1><<<grid, 256, 0, stream>>>(A, Bp, bias, Cf, Cb, M, N, K,
                                                    sA, sB, sC, scale, relu,
                                                    ksplit, atomicAcc);
    else
      wmma_gemm_kernel<0><<<grid, 256, 0, stream>>>(A, Bp, bias, Cf, Cb, M, N, K,
                                                    sA, sB, sC, scale, relu,
                                                    ksplit, atomicAcc);
  };
  auto stats = [&](const float* x, long stride_b, long L) {
    zero(ACC, 2 * B_);
    int gx = (int)((L + 256 * 64 - 1) / (256 * 64));
    if (gx < 1) gx = 1; if (gx > 256) gx = 256;
    region_stats_kernel<<<dim3(gx, B_), 256, 0, stream>>>(x, stride_b, L, ACC);
  };
  auto lnap = [&](const float* x, long stride_b, long L, int relu, float* yf,
                  unsigned short* yb) {
    int gx = (int)((L + 256 * 16 - 1) / (256 * 16));
    if (gx < 1) gx = 1; if (gx > 512) gx = 512;
    ln_apply_kernel<<<dim3(gx, B_), 256, 0, stream>>>(x, stride_b, ACC, L, relu,
                                                      yf, yb);
  };

  // 0) weights -> bf16
  cvt(cW1, W1C, (long)PL_ * DFF_);
  cvt(cW2, W2C, (long)DFF_ * DM_);
  cvt(tW1, W1T, (long)PL_ * DFF_);
  cvt(tW2, W2T, (long)DFF_ * DM_);
  cvt(qW, QW, (long)DM_ * DM_);
  cvt(kW, KW, (long)DM_ * DM_);
  cvt(vW, VW, (long)DM_ * DM_);
  cvt(dW, DW, (long)DM_ * DM_);
  cvt(hW1, HW1, (long)PN_ * DM_ * DFF_);
  cvt(hW2, HW2, (long)DFF_ * PRED_);

  // 1) per-channel normalization + target stats
  channel_norm_kernel<<<B_ * N_, 256, 0, stream>>>(x_enc, XN, TST);

  // 2) cov patch-embed
  stats(XN, (long)N_ * T_, (long)V_ * T_);
  lnap(XN, (long)N_ * T_, (long)V_ * T_, 0, nullptr, A1C);               // ln1
  gemm(A1C, W1C, cb1, C1C, nullptr, B_ * V_ * PN_, DFF_, PL_, 0, 0, 0, 1,
       1.f, 0, 0, 1, 0);                                                 // @W1+b1
  stats(C1C, (long)V_ * PN_ * DFF_, (long)V_ * PN_ * DFF_);
  lnap(C1C, (long)V_ * PN_ * DFF_, (long)V_ * PN_ * DFF_, 1, nullptr, A2C);
  gemm(A2C, W2C, cb2, C2C, nullptr, B_ * V_ * PN_, DM_, DFF_, 0, 0, 0, 1,
       1.f, 0, 0, 1, 0);                                                 // @W2+b2
  stats(C2C, (long)V_ * PN_ * DM_, (long)V_ * PN_ * DM_);
  lnap(C2C, (long)V_ * PN_ * DM_, (long)V_ * PN_ * DM_, 1, nullptr, EXOB);

  // 3) tgt patch-embed
  stats(XN + (long)V_ * T_, (long)N_ * T_, (long)T_);
  lnap(XN + (long)V_ * T_, (long)N_ * T_, (long)T_, 0, nullptr, A1T);
  gemm(A1T, W1T, tb1, C1T, nullptr, B_ * PN_, DFF_, PL_, 0, 0, 0, 1, 1.f, 0, 0,
       1, 0);
  stats(C1T, (long)PN_ * DFF_, (long)PN_ * DFF_);
  lnap(C1T, (long)PN_ * DFF_, (long)PN_ * DFF_, 1, nullptr, A2T);
  gemm(A2T, W2T, tb2, C2T, nullptr, B_ * PN_, DM_, DFF_, 0, 0, 0, 1, 1.f, 0, 0,
       1, 0);
  stats(C2T, (long)PN_ * DM_, (long)PN_ * DM_);
  lnap(C2T, (long)PN_ * DM_, (long)PN_ * DM_, 1, ENDF, ENDB);

  // 4) Q, K, V projections
  gemm(ENDB, QW, qb, nullptr, QB, B_ * PN_, DM_, DM_, 0, 0, 0, 1, 1.f, 0, 0, 1, 0);
  gemm(EXOB, KW, kb, nullptr, KB, B_ * ATTL, DM_, DM_, 0, 0, 0, 1, 1.f, 0, 0, 1, 0);
  gemm(EXOB, VW, vb, nullptr, VB, B_ * ATTL, DM_, DM_, 0, 0, 0, 1, 1.f, 0, 0, 1, 0);

  // 5) scores = Q @ K^T / sqrt(D)  (batched, trans-B)
  gemm(QB, KB, nullptr, SC, nullptr, PN_, ATTL, DM_, (long)PN_ * DM_,
       (long)ATTL * DM_, (long)PN_ * ATTL, B_, 1.f / sqrtf((float)DM_), 0, 1, 1, 0);

  // 6) softmax pair, top-k mask
  softmax_pair_kernel<<<B_ * PN_, 256, 0, stream>>>(SC, ATTN, NATT);
  topk_kernel<<<B_, 256, 0, stream>>>(ATTN, THR, TIES);
  zero(TIEC, B_);
  {
    long tot = (long)B_ * PN_ * ATTL;
    apply_mask_kernel<<<(unsigned)((tot + 255) / 256), 256, 0, stream>>>(
        ATTN, NATT, THR, TIES, TIEC, AIB, AVB);
  }

  // 7) InterI/InterV (batched), dis projection
  gemm(AIB, VB, nullptr, II, IIB, PN_, DM_, ATTL, (long)PN_ * ATTL,
       (long)ATTL * DM_, (long)PN_ * DM_, B_, 1.f, 0, 0, 1, 0);
  gemm(AVB, VB, nullptr, IV, nullptr, PN_, DM_, ATTL, (long)PN_ * ATTL,
       (long)ATTL * DM_, (long)PN_ * DM_, B_, 1.f, 0, 0, 1, 0);
  gemm(IIB, DW, db, MD, nullptr, B_ * PN_, DM_, DM_, 0, 0, 0, 1, 1.f, 0, 0, 1, 0);

  // 8) gating + three head inputs
  {
    long tot = (long)B_ * PN_ * DM_;
    fuse_gate_kernel<<<(unsigned)((tot + 255) / 256), 256, 0, stream>>>(
        II, IV, MD, ENDF, alpha, H);
  }

  // 9) head: split-K GEMM1 (K=32768) with atomic accumulate, row-LN+relu, GEMM2
  zero(E1, (long)3 * B_ * DFF_);
  gemm(H, HW1, nullptr, E1, nullptr, 3 * B_, DFF_, PN_ * DM_, 0, 0, 0, 1, 1.f, 0,
       0, 16, 1);
  head_ln_kernel<<<3 * B_, 256, 0, stream>>>(E1, hb1, hlg, hlb, E1B);
  gemm(E1B, HW2, hb2, E2, nullptr, 3 * B_, PRED_, DFF_, 0, 0, 0, 1, 1.f, 0, 0, 1, 0);

  // 10) de-normalize -> (3, B, PRED, 1)
  {
    long tot = (long)3 * B_ * PRED_;
    final_out_kernel<<<(unsigned)((tot + 255) / 256), 256, 0, stream>>>(
        E2, TST, (float*)d_out);
  }
}
